// svdconv_66632122630494
// MI455X (gfx1250) — matrix operand order repaired and checked by reading, not compiled, not run on hardware
//
#include <hip/hip_runtime.h>

// SVD-factorized 3x3 conv, fused two-stage WMMA kernel for gfx1250 (wave32).
// x:(32,256,56,56) f32, U:(3,3,256,64) f32, V:(3,3,64,256) f32 -> out:(32,256,56,56) f32
// f32 -> f16 operands, v_wmma_f32_16x16x32_f16, f32 accumulate.
// - Prelude pre-converts U/V to f16 in WMMA A-fragment lane order (d_ws).
// - 9-wave workgroups: one conv tap per wave; each A fragment reused across
//   two 16-pixel groups (N=32/workgroup) to halve L2 weight-fragment traffic.
// - x halo staged in two 128-channel LDS chunks to fit 64 KB alongside t.

#define Bsz 32
#define Cch 256
#define Hh  56
#define Ww  56
#define OCc 256
#define Rr  64
#define HW  (Hh * Ww)

#define TILE_H 512            // swizzled A tile: 32 lanes x 16 f16 = 1024 B
#define NVTILE (9 * 8 * 4)    // [tap][kc=8][m=4]
#define NUTILE (9 * 2 * 16)   // [tap][kc=2][ms=16]

typedef __attribute__((ext_vector_type(16))) _Float16 v16h;
typedef __attribute__((ext_vector_type(8)))  _Float16 v8h;
typedef __attribute__((ext_vector_type(8)))  float    v8f;

__device__ inline v16h ld_frag(const _Float16* __restrict__ p) {  // global, lane*32B
  union { v16h v; v8h h[2]; } u;
  u.h[0] = *(const v8h*)(p);
  u.h[1] = *(const v8h*)(p + 8);
  return u.v;
}

__device__ inline v16h ld_b16(const _Float16* p) {  // LDS, 2x ds_load_b128
  union { v16h v; v8h h[2]; } u;
  u.h[0] = *(const v8h*)(p);
  u.h[1] = *(const v8h*)(p + 8);
  return u.v;
}

__device__ inline void st_t8(_Float16* p, v8f a) {  // pack 8 f32 -> 8 f16, b128 store
  union { v8h h; _Float16 e[8]; } u;
#pragma unroll
  for (int i = 0; i < 8; ++i) u.e[i] = (_Float16)a[i];
  *(v8h*)p = u.h;
}

__device__ inline v8f wmma_f16(v16h a, v16h b, v8f c) {
  return __builtin_amdgcn_wmma_f32_16x16x32_f16(false, a, false, b,
                                                (short)0, c, false, false);
}

// ---- Prelude: U/V f32 -> f16 in A-fragment lane order ---------------------
// A fragment (16x32 f16): lane L = h2*16 + mrow holds row M=mrow,
// K = {16*h2+0..15} of the 32-wide K chunk.
__global__ __launch_bounds__(256) void swizzle_weights_kernel(
    const float* __restrict__ U, const float* __restrict__ V,
    _Float16* __restrict__ vsw, _Float16* __restrict__ usw) {
  const int wv   = (blockIdx.x * 8) + (threadIdx.x >> 5);
  const int lane = threadIdx.x & 31;
  const int h2   = lane >> 4;
  const int mrow = lane & 15;
  if (wv >= NVTILE + NUTILE) return;

  const float* src;
  _Float16* dst;
  if (wv < NVTILE) {
    const int m   = wv & 3;
    const int kc  = (wv >> 2) & 7;
    const int tap = wv >> 5;
    src = V + ((size_t)(tap * Rr + m * 16 + mrow)) * Cch + kc * 32 + 16 * h2;
    dst = vsw + (size_t)wv * TILE_H + lane * 16;
  } else {
    const int u   = wv - NVTILE;
    const int ms  = u & 15;
    const int kc  = (u >> 4) & 1;
    const int tap = u >> 5;
    src = U + ((size_t)(tap * OCc + ms * 16 + mrow)) * Rr + kc * 32 + 16 * h2;
    dst = usw + (size_t)u * TILE_H + lane * 16;
  }
  union { v8h h[2]; _Float16 e[16]; } o;
#pragma unroll
  for (int i = 0; i < 8; ++i) {
    o.e[i]     = (_Float16)src[i];
    o.e[8 + i] = (_Float16)src[8 + i];
  }
  *(v8h*)(dst)     = o.h[0];
  *(v8h*)(dst + 8) = o.h[1];
}

// ---- Main fused kernel: 288 threads = 9 waves, 32 px x 256 OC per WG ------
__global__ __launch_bounds__(288) void svdconv_fused_kernel(
    const float* __restrict__ x, const _Float16* __restrict__ vsw,
    const _Float16* __restrict__ usw, float* __restrict__ out) {
  __shared__ __align__(16) _Float16 xs[3 * 34][128];   // 26112 B halo, 128-ch chunk
  __shared__ __align__(16) _Float16 ts[9][32][Rr];     // 36864 B per-tap t (32 px)

  const int tid  = threadIdx.x;
  const int wave = tid >> 5;       // 0..8 == conv tap
  const int lane = tid & 31;
  const int h2   = lane >> 4;
  const int ln   = lane & 15;

  const int gid = blockIdx.x;
  const int gi  = gid & 1;                  // row half: px [0,32) or [24,56)
  const int h0  = (gid >> 1) % Hh;
  const int b   = (gid >> 1) / Hh;
  const int wbase = gi * 24;                // overlap px 24..31: dup writes benign

  // Stage-1 setup: one tap per wave.
  const int tap = wave;
  const int di  = tap / 3 - 1;
  const int dj  = tap % 3 - 1;
  const int q0  = (1 + di) * 34 + (ln + 1 + dj);        // halo px, group 0
  const int q1  = q0 + 16;                              // halo px, group 1
  const _Float16* vt = vsw + ((size_t)tap * 32) * TILE_H + lane * 16;

  v8f acc[2][4] = {{{}, {}, {}, {}}, {{}, {}, {}, {}}};

#pragma unroll
  for (int chunk = 0; chunk < 2; ++chunk) {
    if (chunk) __syncthreads();   // everyone done reading previous chunk
    // Stage x halo chunk (3 rows x 34 px x 128 ch) as f16, zero padding.
    for (int e = tid; e < 3 * 128 * 34; e += 288) {
      const int wi  = e % 34;
      const int rc  = e / 34;
      const int cl  = rc & 127;
      const int row = rc >> 7;
      const int hh  = h0 - 1 + row;
      const int ww  = wbase - 1 + wi;
      float v = 0.0f;
      if ((unsigned)hh < (unsigned)Hh && (unsigned)ww < (unsigned)Ww)
        v = x[(((size_t)b * Cch + chunk * 128 + cl) * Hh + hh) * Ww + ww];
      xs[row * 34 + wi][cl] = (_Float16)v;
    }
    __syncthreads();

    // Stage 1: t_tap(64 x 32px) += V_tap(64x128) * x_shift(128x32); each V
    // fragment reused for both pixel groups. EXEC all-ones (uniform flow).
#pragma unroll
    for (int kc4 = 0; kc4 < 4; ++kc4) {
      const v16h b0 = ld_b16(&xs[q0][kc4 * 32 + 16 * h2]);
      const v16h b1 = ld_b16(&xs[q1][kc4 * 32 + 16 * h2]);
      const int kc = chunk * 4 + kc4;
#pragma unroll
      for (int m = 0; m < 4; ++m) {
        const v16h am = ld_frag(vt + (size_t)(kc * 4 + m) * TILE_H);
        acc[0][m] = wmma_f16(am, b0, acc[0][m]);
        acc[1][m] = wmma_f16(am, b1, acc[1][m]);
      }
    }
  }

  // D-layout -> [pixel][R] f16 in LDS (tap-private region).
#pragma unroll
  for (int g = 0; g < 2; ++g)
#pragma unroll
    for (int m = 0; m < 4; ++m)
      st_t8(&ts[tap][g * 16 + ln][m * 16 + 8 * h2], acc[g][m]);
  __syncthreads();

  // Stage 2: out(256 x 32px) = sum_tap U_tap(256x64) * t_tap(64x32px).
  // Wave owns whole OC subtiles; each U fragment reused for both groups.
  const int p0 = h0 * Ww + wbase;
  for (int ms = wave; ms < 16; ms += 9) {
    const _Float16* ut = usw + (size_t)ms * TILE_H + lane * 16;
    v8f oacc[2] = {{}, {}};
#pragma unroll
    for (int tp = 0; tp < 9; ++tp) {
#pragma unroll
      for (int kc = 0; kc < 2; ++kc) {
        const v16h am = ld_frag(ut + (size_t)((tp * 2 + kc) * 16) * TILE_H);
#pragma unroll
        for (int g = 0; g < 2; ++g) {
          const v16h bm = ld_b16(&ts[tp][g * 16 + ln][kc * 32 + 16 * h2]);
          oacc[g] = wmma_f16(am, bm, oacc[g]);
        }
      }
    }
#pragma unroll
    for (int g = 0; g < 2; ++g) {
      float* op = out + (((size_t)b * OCc + ms * 16 + 8 * h2) * HW) + p0 + g * 16 + ln;
#pragma unroll
      for (int vr = 0; vr < 8; ++vr)
        op[(size_t)vr * HW] = oacc[g][vr];
    }
  }
}

extern "C" void kernel_launch(void* const* d_in, const int* in_sizes, int n_in,
                              void* d_out, int out_size, void* d_ws, size_t ws_size,
                              hipStream_t stream) {
  const float* x = (const float*)d_in[0];
  const float* U = (const float*)d_in[1];
  const float* V = (const float*)d_in[2];
  float* out = (float*)d_out;
  (void)in_sizes; (void)n_in; (void)out_size; (void)ws_size;

  _Float16* vsw = (_Float16*)d_ws;                 // 294912 B
  _Float16* usw = vsw + (size_t)NVTILE * TILE_H;   // 294912 B (total < 0.6 MB)

  swizzle_weights_kernel<<<(NVTILE + NUTILE + 7) / 8, 256, 0, stream>>>(U, V, vsw, usw);

  dim3 grid(Bsz * Hh * 2);   // 3584 workgroups, 32 px x 256 OC each
  dim3 block(288);           // 9 waves: one conv tap per wave
  svdconv_fused_kernel<<<grid, block, 0, stream>>>(x, vsw, usw, out);
}
